// Attention_83803401879967
// MI455X (gfx1250) — compile-verified, matrix operand order checked by reading, not run on hardware
//
#include <hip/hip_runtime.h>
#include <hip/hip_bf16.h>

typedef __attribute__((ext_vector_type(16))) __bf16 v16bf;
typedef __attribute__((ext_vector_type(8)))  float  v8f;

#define E_DIM 2048
#define D_DIM 1024
#define A_DIM 1024
#define BATCH 64
#define N_DIM 1024

#if __has_builtin(__builtin_amdgcn_sched_barrier)
#define SCHED_FENCE() __builtin_amdgcn_sched_barrier(0)
#else
#define SCHED_FENCE()
#endif

__device__ __forceinline__ unsigned short f2bf(float f) {
  union { float f; unsigned int u; } c; c.f = f;
  unsigned int u = c.u;
  return (unsigned short)((u + 0x7FFFu + ((u >> 16) & 1u)) >> 16); // RNE
}

union Frag {
  v16bf v;
  uint4 q[2];
};

#define WMMA_BF16(A, B, C) \
  __builtin_amdgcn_wmma_f32_16x16x32_bf16(false, (A).v, false, (B).v, (short)0, (C), false, false)

// ---------------------------------------------------------------------------
// Kernel 1: W_enc [E,A] f32 -> Wt [A,E] bf16 (transposed so B-fragments are
// contiguous 16B loads, and the 4 MB result stays L2-resident).
// ---------------------------------------------------------------------------
__global__ void k_convW(const float* __restrict__ W_enc,
                        unsigned short* __restrict__ Wt) {
  int idx = blockIdx.x * blockDim.x + threadIdx.x;   // over A*E, k-fastest out
  int a = idx >> 11;          // idx / E_DIM
  int k = idx & (E_DIM - 1);
  Wt[idx] = f2bf(W_enc[k * A_DIM + a]);
}

// ---------------------------------------------------------------------------
// Kernel 2: comb[b][a] = dec_out[b,:] . W_dec[:,a] + b_dec[a] + b_enc[a]
// (b_att omitted: softmax(x + c) == softmax(x), so it cannot affect outputs)
// ---------------------------------------------------------------------------
__global__ void k_comb(const float* __restrict__ dec_out,
                       const float* __restrict__ W_dec,
                       const float* __restrict__ b_dec,
                       const float* __restrict__ b_enc,
                       float* __restrict__ comb) {
  int a = blockIdx.x * blockDim.x + threadIdx.x;   // 0..1023
  int b = blockIdx.y;
  float acc = b_dec[a] + b_enc[a];
  const float* dp = dec_out + b * D_DIM;
  for (int k = 0; k < D_DIM; ++k)
    acc = fmaf(dp[k], W_dec[k * A_DIM + a], acc);   // coalesced across a
  comb[b * A_DIM + a] = acc;
}

// ---------------------------------------------------------------------------
// Kernel 3 (main): fused enc@W_enc -> relu(+comb) -> dot(W_att) -> scores.
// 64 enc rows pinned in LDS as bf16 (256 KB of the 320 KB WGP LDS); 16 waves
// sweep all 1024 A-columns in 4 chunks with f32 WMMA accumulators.
// Software pipeline: B fragment (global) double-buffered one K-step ahead;
// all 4 A fragments loaded before the 4 WMMAs with a sched_barrier between
// the load group and the WMMA group, so the fragments stay in distinct
// registers (no WMMA WAR hazard v_nops) and dep-counter waits are staggered.
// ---------------------------------------------------------------------------
__global__ __launch_bounds__(512)
void k_scores(const float* __restrict__ enc_out,
              const unsigned short* __restrict__ Wt,
              const float* __restrict__ comb,
              const float* __restrict__ W_att,
              float* __restrict__ scores) {
  extern __shared__ unsigned short encLds[];   // 64 * 2048 bf16 = 256 KB
  __shared__ float scoreLds[64];

  const int tid  = threadIdx.x;
  const int wave = tid >> 5;
  const int lane = tid & 31;
  const int r0   = blockIdx.x * 64;            // global row base (64 | 1024)
  const int b    = r0 >> 10;                   // batch index

  // ---- Phase 1: stage 64x2048 enc tile to LDS as bf16 (float4 reads) ----
  const float* ep = enc_out + (size_t)r0 * E_DIM;
  for (int i = tid; i < (64 * E_DIM) / 4; i += 512) {
    const float4 v = ((const float4*)ep)[i];
    uint2 p;
    p.x = (unsigned)f2bf(v.x) | ((unsigned)f2bf(v.y) << 16);
    p.y = (unsigned)f2bf(v.z) | ((unsigned)f2bf(v.w) << 16);
    *(uint2*)(&encLds[i * 4]) = p;
  }
  if (tid < 64) scoreLds[tid] = 0.0f;
  __syncthreads();

  const int halfSel = lane >> 4;               // K-half select per WMMA layout
  const int lane15  = lane & 15;
  const unsigned short* aBase = encLds + lane15 * E_DIM + halfSel * 8;

  for (int chunk = 0; chunk < 4; ++chunk) {
    const int col = chunk * 256 + wave * 16 + lane15;   // this lane's A column
    const float cb = comb[b * A_DIM + col];
    const float wa = W_att[col];
    const unsigned short* wbase = Wt + (size_t)col * E_DIM + halfSel * 8;

    auto loadB = [&](Frag& f, int k) {
      f.q[0] = *(const uint4*)(wbase + k);
      f.q[1] = *(const uint4*)(wbase + k + 16);
    };
    auto loadA = [&](Frag& f, int t, int k) {
      const unsigned short* p = aBase + t * 16 * E_DIM + k;
      f.q[0] = *(const uint4*)(p);
      f.q[1] = *(const uint4*)(p + 16);
    };

    v8f acc0 = {}, acc1 = {}, acc2 = {}, acc3 = {};

    Frag b0, b1;
    loadB(b0, 0);                               // prologue: first B fragment
    for (int k0 = 0; k0 < E_DIM; k0 += 64) {
      // --- step 0: consume b0 @ K=k0, prefetch b1 @ K=k0+32 ---
      {
        Frag fa0, fa1, fa2, fa3;
        loadB(b1, k0 + 32);
        loadA(fa0, 0, k0);
        loadA(fa1, 1, k0);
        loadA(fa2, 2, k0);
        loadA(fa3, 3, k0);
        SCHED_FENCE();                 // keep loads above, frags in distinct regs
        acc0 = WMMA_BF16(fa0, b0, acc0);
        acc1 = WMMA_BF16(fa1, b0, acc1);
        acc2 = WMMA_BF16(fa2, b0, acc2);
        acc3 = WMMA_BF16(fa3, b0, acc3);
      }
      // --- step 1: consume b1 @ K=k0+32, prefetch b0 @ next (wrap-safe) ---
      {
        const int k1 = k0 + 32;
        Frag fa0, fa1, fa2, fa3;
        loadB(b0, (k0 + 64) & (E_DIM - 1));
        loadA(fa0, 0, k1);
        loadA(fa1, 1, k1);
        loadA(fa2, 2, k1);
        loadA(fa3, 3, k1);
        SCHED_FENCE();
        acc0 = WMMA_BF16(fa0, b1, acc0);
        acc1 = WMMA_BF16(fa1, b1, acc1);
        acc2 = WMMA_BF16(fa2, b1, acc2);
        acc3 = WMMA_BF16(fa3, b1, acc3);
      }
    }

    // ---- Epilogue: relu(acc+comb)*W_att, reduce 16 cols per half-wave ----
    auto reduce_tile = [&](const v8f acc, int t) {
#pragma unroll
      for (int j = 0; j < 8; ++j) {
        float v = fmaxf(acc[j] + cb, 0.0f) * wa;
        v += __shfl_xor(v, 1, 32);   // xor offsets <16 stay within half-wave
        v += __shfl_xor(v, 2, 32);
        v += __shfl_xor(v, 4, 32);
        v += __shfl_xor(v, 8, 32);
        if (lane15 == 0)
          atomicAdd(&scoreLds[t * 16 + j + halfSel * 8], v);  // ds_add_f32
      }
    };
    reduce_tile(acc0, 0);
    reduce_tile(acc1, 1);
    reduce_tile(acc2, 2);
    reduce_tile(acc3, 3);
  }

  __syncthreads();
  if (tid < 64) scores[r0 + tid] = scoreLds[tid];
}

// ---------------------------------------------------------------------------
// Kernel 4: per-batch softmax over N=1024 -> alpha
// ---------------------------------------------------------------------------
__global__ void k_softmax(const float* __restrict__ scores,
                          float* __restrict__ alpha) {
  __shared__ float red[256];
  const int b = blockIdx.x, tid = threadIdx.x;
  const float* sp = scores + b * N_DIM;
  float x0 = sp[tid], x1 = sp[tid + 256], x2 = sp[tid + 512], x3 = sp[tid + 768];

  float m = fmaxf(fmaxf(x0, x1), fmaxf(x2, x3));
  red[tid] = m; __syncthreads();
  for (int s = 128; s > 0; s >>= 1) {
    if (tid < s) red[tid] = fmaxf(red[tid], red[tid + s]);
    __syncthreads();
  }
  m = red[0]; __syncthreads();

  float e0 = expf(x0 - m), e1 = expf(x1 - m), e2 = expf(x2 - m), e3 = expf(x3 - m);
  red[tid] = e0 + e1 + e2 + e3; __syncthreads();
  for (int s = 128; s > 0; s >>= 1) {
    if (tid < s) red[tid] += red[tid + s];
    __syncthreads();
  }
  const float inv = 1.0f / red[0];
  float* ap = alpha + b * N_DIM;
  ap[tid] = e0 * inv; ap[tid + 256] = e1 * inv;
  ap[tid + 512] = e2 * inv; ap[tid + 768] = e3 * inv;
}

// ---------------------------------------------------------------------------
// Kernel 5: weighted_enc[b,e] = sum_n alpha[b,n] * enc_out[b,n,e]
// ---------------------------------------------------------------------------
__global__ void k_weighted(const float* __restrict__ enc_out,
                           const float* __restrict__ alpha,
                           float* __restrict__ out) {
  __shared__ float al[N_DIM];
  const int b = blockIdx.y;
  const int e = blockIdx.x * 256 + threadIdx.x;
  for (int i = threadIdx.x; i < N_DIM; i += 256) al[i] = alpha[b * N_DIM + i];
  __syncthreads();
  const float* ep = enc_out + (size_t)b * N_DIM * E_DIM + e;
  float acc = 0.0f;
  for (int n = 0; n < N_DIM; ++n)
    acc = fmaf(al[n], ep[(size_t)n * E_DIM], acc);  // coalesced across e
  out[b * E_DIM + e] = acc;
}

// ---------------------------------------------------------------------------
extern "C" void kernel_launch(void* const* d_in, const int* in_sizes, int n_in,
                              void* d_out, int out_size, void* d_ws, size_t ws_size,
                              hipStream_t stream) {
  const float* enc_out = (const float*)d_in[0];
  const float* dec_out = (const float*)d_in[1];
  const float* W_enc   = (const float*)d_in[2];
  const float* b_enc   = (const float*)d_in[3];
  const float* W_dec   = (const float*)d_in[4];
  const float* b_dec   = (const float*)d_in[5];
  const float* W_att   = (const float*)d_in[6];
  // d_in[7] = b_att: softmax-invariant, intentionally unused.

  char* ws = (char*)d_ws;
  unsigned short* Wt = (unsigned short*)ws;                              // 4 MB
  float* comb   = (float*)(ws + (size_t)A_DIM * E_DIM * sizeof(unsigned short));
  float* scores = comb + BATCH * A_DIM;                                   // +256 KB

  float* weighted = (float*)d_out;                 // [64, 2048]
  float* alpha    = weighted + BATCH * E_DIM;      // [64, 1024]

  k_convW<<<(A_DIM * E_DIM) / 256, 256, 0, stream>>>(W_enc, Wt);
  k_comb<<<dim3(A_DIM / 256, BATCH), 256, 0, stream>>>(dec_out, W_dec, b_dec, b_enc, comb);
  k_scores<<<(BATCH * N_DIM) / 64, 512, 64 * E_DIM * sizeof(unsigned short), stream>>>(
      enc_out, Wt, comb, W_att, scores);
  k_softmax<<<BATCH, 256, 0, stream>>>(scores, alpha);
  k_weighted<<<dim3(E_DIM / 256, BATCH), 256, 0, stream>>>(enc_out, alpha, weighted);
}
